// Argphy_55430847922306
// MI455X (gfx1250) — matrix operand order repaired and checked by reading, not compiled, not run on hardware
//
#include <hip/hip_runtime.h>
#include <hip/hip_bf16.h>

typedef __attribute__((ext_vector_type(16))) _Float16 v16h;
typedef __attribute__((ext_vector_type(8)))  float    v8f;

#define HIDN 256
#define WIDN 512
#define TN   512
#define SUBN 4

// RK coefficients
#define A21f 0.161f
#define A31f (-0.008480655492356989f)
#define A32f 0.335480655492357f
#define A41f 2.8971530571054935f
#define A42f (-6.359448489975075f)
#define A43f 4.3622954328695815f
#define A51f 5.325864828439257f
#define A52f (-11.748883564062828f)
#define A53f 7.4955393428898365f
#define A54f (-0.09249506636175525f)
#define A61f 5.86145544294642f
#define A62f (-12.92096931784711f)
#define A63f 8.159367898576159f
#define A64f (-0.071584973281401f)
#define A65f (-0.028269050394068383f)
#define B1f 0.09646076681806523f
#define B2f 0.01f
#define B3f 0.4798896504144996f
#define B4f 1.379008574103742f
#define B5f (-3.290069515436081f)
#define B6f 2.324710524099774f

__device__ __forceinline__ float softplusf(float v) {
    return v > 20.0f ? v : log1pf(expf(v));
}

// ---------------------------------------------------------------------------
// Prep kernel: repack W1/W2/W3 (fp32 row-major [out,in]) into f16 WMMA
// A-fragments, K contiguous within each lane half (this code's order).
//   frag global index: L1: mt*8+kt (256 frags), L2: 256 + mt*16+kt (512),
//                      L3: 768 + mt*16+kt (256).  Each frag = 32 lanes * 16 f16.
//   lane l<16  : row = mt*16+l,      K = kt*32 + {0..15}
//   lane l>=16 : row = mt*16+(l-16), K = kt*32 + {16..31}
// ---------------------------------------------------------------------------
__global__ void prep_kernel(const float* __restrict__ W1,
                            const float* __restrict__ W2,
                            const float* __restrict__ W3,
                            _Float16* __restrict__ frags) {
    int slot = blockIdx.x * blockDim.x + threadIdx.x;   // 0..32767
    int fl   = slot >> 5;
    int lane = slot & 31;
    const float* W;
    int IN, mt, kt;
    if (fl < 256)      { W = W1; IN = 256; mt = fl >> 3;        kt = fl & 7;  }
    else if (fl < 768) { W = W2; IN = 512; int f = fl - 256; mt = f >> 4; kt = f & 15; }
    else               { W = W3; IN = 512; int f = fl - 768; mt = f >> 4; kt = f & 15; }
    int row  = mt * 16 + (lane & 15);
    int half = lane >> 4;
    _Float16* dst = frags + (size_t)slot * 16;
#pragma unroll
    for (int e = 0; e < 16; ++e) {
        int k = kt * 32 + half * 16 + e;
        dst[e] = (_Float16)W[(size_t)row * IN + k];
    }
}

// ---------------------------------------------------------------------------
// Persistent single-workgroup ODE integrator. 1024 threads = 32 wave32s.
// All state lives in LDS; weights streamed (L2-resident) as WMMA fragments.
// ---------------------------------------------------------------------------
__global__ __launch_bounds__(1024)
void ode_kernel(const float* __restrict__ ts,
                const float* __restrict__ scaleP,
                const float* __restrict__ b1,
                const float* __restrict__ b2,
                const float* __restrict__ b3,
                const float* __restrict__ htbW,
                const float* __restrict__ htbB,
                const float* __restrict__ hidden,
                const float* __restrict__ statev,
                const float* __restrict__ scalesP,
                const _Float16* __restrict__ frags,
                float* __restrict__ out) {
    __shared__ float    y[261];
    __shared__ float    ytmp[261];
    __shared__ float    ks[6][261];
    __shared__ _Float16 xin_h[HIDN];   // f16 MLP input h   (B operand, layer 1)
    __shared__ _Float16 act_h[WIDN];   // f16 activations   (B operand, layers 2/3)
    __shared__ float    zraw[WIDN];    // raw GEMV output
    __shared__ float    z3a[HIDN];     // layer3 split-K partials
    __shared__ float    z3b[HIDN];
    __shared__ float    b1s[WIDN], b2s[WIDN], b3s[HIDN], htbWs[HIDN];
    __shared__ float    sc[5];
    __shared__ float    sdt_sh;

    const int t    = threadIdx.x;
    const int w    = t >> 5;
    const int lane = t & 31;
    const float scaleV = scaleP[0];
    const float htbB0  = htbB[0];

    // One-time LDS caching of small constants
    if (t < 5) sc[t] = scalesP[t];
    if (t < WIDN) { b1s[t] = b1[t]; b2s[t] = b2[t]; }
    if (t >= WIDN) { int u = t - WIDN; if (u < HIDN) b3s[u] = b3[u]; else htbWs[u - HIDN] = htbW[u - HIDN]; }
    __syncthreads();

    // GEMV via WMMA: D(:,0) = W_tile(16xK) @ x.
    // Only D column 0 is consumed, so B columns 1..15 may hold garbage:
    // every lane loads 32B of x from LDS, no masking, no divergence.
    auto gemv = [&](int fragBase, int nkt, const _Float16* xh, float* zout, int mt) {
        v8f acc = {};
        const int boff = lane & 16;   // lanes 0-15 -> K 0..15, lanes 16-31 -> K 16..31
#pragma unroll 4
        for (int kt = 0; kt < nkt; ++kt) {
            const v16h* ap = (const v16h*)(frags + (size_t)(fragBase + kt) * 512);
            v16h a = ap[lane];
            v16h b = *(const v16h*)(xh + kt * 32 + boff);
            acc = __builtin_amdgcn_wmma_f32_16x16x32_f16(
                false, a, false, b, (short)0, acc, false, false);
        }
        if ((lane & 15) == 0) {       // lanes 0 and 16 hold column N=0
            float* dst = zout + mt * 16 + (lane >> 4) * 8;
#pragma unroll
            for (int r = 0; r < 8; ++r) dst[r] = acc[r];
        }
    };

    // Precondition: caller issued a barrier after writing yv[] and xin_h[].
    // Postcondition: kout fully written; caller must barrier before reading.
    auto rhs = [&](const float* yv, float* kout) {
        // ---- beta + epidemic derivatives: wave 0 only, shuffle reduction,
        //      runs concurrently with other waves' layer-1 WMMAs.
        if (w == 0) {
            float p = 0.0f;
#pragma unroll
            for (int j = 0; j < 8; ++j)
                p += htbWs[lane + 32 * j] * yv[5 + lane + 32 * j];
#pragma unroll
            for (int m = 16; m >= 1; m >>= 1) p += __shfl_xor(p, m, 32);
            if (lane == 0) {
                float beta = 1.0f / (1.0f + expf(-(p + htbB0)));
                float S = yv[0] * sc[0], E = yv[1] * sc[1], I = yv[2] * sc[2];
                float A = yv[3] * sc[3];
                const float kkc = 0.526f, aac = 0.244f, iic = 0.244f, pc = 0.667f;
                const float fc = 0.98f, qc = 0.5f;
                float LL = (1.0f - qc) * I + A;     // ee=0, dd=1
                float dS = -beta * S * LL;
                float dE = beta * S * LL - kkc * E;
                float dI = pc * kkc * E - aac * I;
                float dA = (1.0f - pc) * kkc * E - iic * A;
                float dR = fc * aac * I + iic * A;
                kout[0] = dS / sc[0]; kout[1] = dE / sc[1]; kout[2] = dI / sc[2];
                kout[3] = dA / sc[3]; kout[4] = dR / sc[4];
            }
        }
        // ---- layer 1: 512x256, all 32 waves, 8 K-tiles each
        gemv(w * 8, 8, xin_h, zraw, w);
        __syncthreads();
        if (t < WIDN) act_h[t] = (_Float16)softplusf(zraw[t] + b1s[t]);
        __syncthreads();
        // ---- layer 2: 512x512, all 32 waves, 16 K-tiles each
        gemv(256 + w * 16, 16, act_h, zraw, w);
        __syncthreads();
        if (t < WIDN) act_h[t] = (_Float16)softplusf(zraw[t] + b2s[t]);
        __syncthreads();
        // ---- layer 3: 256x512, split-K across wave halves
        if (w < 16) gemv(768 + w * 16, 8, act_h, z3a, w);
        else        gemv(768 + (w - 16) * 16 + 8, 8, act_h + 256, z3b, w - 16);
        __syncthreads();
        if (t < HIDN)
            kout[5 + t] = scaleV * tanhf(1e-4f * (z3a[t] + z3b[t] + b3s[t]));
    };

    auto saveRow = [&](int row) {
        if (t < 5)    out[(size_t)row * 5 + t] = y[t] * sc[t];
        if (t < HIDN) out[(size_t)TN * 5 + (size_t)row * HIDN + t] = y[5 + t];
    };

    // ---- initial state: y0 = [softmax(state_vec)/scales, hidden_vec]
    if (t == 0) {
        float m = statev[0];
        for (int i = 1; i < 5; ++i) m = fmaxf(m, statev[i]);
        float e[5], ssum = 0.0f;
        for (int i = 0; i < 5; ++i) { e[i] = expf(statev[i] - m); ssum += e[i]; }
        for (int i = 0; i < 5; ++i) y[i] = (e[i] / ssum) / sc[i];
    }
    if (t < HIDN) { float v = hidden[t]; y[5 + t] = v; xin_h[t] = (_Float16)v; }
    __syncthreads();
    saveRow(0);

    int sglob = 0;
    for (int i = 0; i < TN - 1; ++i) {
        if (t == 0) sdt_sh = (ts[i + 1] - ts[i]) * (1.0f / SUBN);
        __syncthreads();
        const float dt = sdt_sh;
        for (int sub = 0; sub < SUBN; ++sub) {
            rhs(y, ks[0]);
            __syncthreads();
            if (t < 261) {
                float v = y[t] + dt * (A21f * ks[0][t]);
                ytmp[t] = v; if (t >= 5) xin_h[t - 5] = (_Float16)v;
            }
            __syncthreads();
            rhs(ytmp, ks[1]);
            __syncthreads();
            if (t < 261) {
                float v = y[t] + dt * (A31f * ks[0][t] + A32f * ks[1][t]);
                ytmp[t] = v; if (t >= 5) xin_h[t - 5] = (_Float16)v;
            }
            __syncthreads();
            rhs(ytmp, ks[2]);
            __syncthreads();
            if (t < 261) {
                float v = y[t] + dt * (A41f * ks[0][t] + A42f * ks[1][t] + A43f * ks[2][t]);
                ytmp[t] = v; if (t >= 5) xin_h[t - 5] = (_Float16)v;
            }
            __syncthreads();
            rhs(ytmp, ks[3]);
            __syncthreads();
            if (t < 261) {
                float v = y[t] + dt * (A51f * ks[0][t] + A52f * ks[1][t] +
                                       A53f * ks[2][t] + A54f * ks[3][t]);
                ytmp[t] = v; if (t >= 5) xin_h[t - 5] = (_Float16)v;
            }
            __syncthreads();
            rhs(ytmp, ks[4]);
            __syncthreads();
            if (t < 261) {
                float v = y[t] + dt * (A61f * ks[0][t] + A62f * ks[1][t] +
                                       A63f * ks[2][t] + A64f * ks[3][t] + A65f * ks[4][t]);
                ytmp[t] = v; if (t >= 5) xin_h[t - 5] = (_Float16)v;
            }
            __syncthreads();
            rhs(ytmp, ks[5]);
            __syncthreads();
            if (t < 261) {
                float v = y[t] + dt * (B1f * ks[0][t] + B2f * ks[1][t] + B3f * ks[2][t] +
                                       B4f * ks[3][t] + B5f * ks[4][t] + B6f * ks[5][t]);
                y[t] = v; if (t >= 5) xin_h[t - 5] = (_Float16)v;
            }
            __syncthreads();
            ++sglob;
            if ((sglob & 3) == 0) saveRow(sglob >> 2);
        }
    }
}

extern "C" void kernel_launch(void* const* d_in, const int* in_sizes, int n_in,
                              void* d_out, int out_size, void* d_ws, size_t ws_size,
                              hipStream_t stream) {
    (void)in_sizes; (void)n_in; (void)out_size; (void)ws_size;
    const float* ts     = (const float*)d_in[1];
    const float* scale  = (const float*)d_in[2];
    const float* W1     = (const float*)d_in[3];
    const float* b1     = (const float*)d_in[4];
    const float* W2     = (const float*)d_in[5];
    const float* b2     = (const float*)d_in[6];
    const float* W3     = (const float*)d_in[7];
    const float* b3     = (const float*)d_in[8];
    const float* htbW   = (const float*)d_in[9];
    const float* htbB   = (const float*)d_in[10];
    const float* hidden = (const float*)d_in[11];
    const float* statev = (const float*)d_in[12];
    const float* scales = (const float*)d_in[13];
    _Float16* frags = (_Float16*)d_ws;   // 1024 frags * 512 f16 = 1 MB

    prep_kernel<<<128, 256, 0, stream>>>(W1, W2, W3, frags);
    ode_kernel<<<1, 1024, 0, stream>>>(ts, scale, b1, b2, b3, htbW, htbB,
                                       hidden, statev, scales, frags,
                                       (float*)d_out);
}
